// SelfAttentionBlock_70231305224711
// MI455X (gfx1250) — compile-verified, matrix operand order checked by reading, not run on hardware
//
#include <hip/hip_runtime.h>

typedef __attribute__((ext_vector_type(16))) __bf16 v16bf;
typedef __attribute__((ext_vector_type(8)))  float  v8f;

constexpr int BATCH = 4, S = 2048, C = 512, H = 8, D = 64, HD = 512;
constexpr int ROWS = BATCH * S;

__device__ __forceinline__ unsigned short f2bf(float x) {
  unsigned int u = __float_as_uint(x);
  u += 0x7FFFu + ((u >> 16) & 1u);          // round-to-nearest-even
  return (unsigned short)(u >> 16);
}

__device__ __forceinline__ v8f wmma_bf16(v16bf a, v16bf b, v8f c) {
  // D = A(16x32 bf16) * B(32x16 bf16) + C(f32)
  return __builtin_amdgcn_wmma_f32_16x16x32_bf16(false, a, false, b,
                                                 (short)0, c, false, false);
}

// Low 32 bits of a generic pointer to __shared__ == LDS byte offset
__device__ __forceinline__ unsigned ldsOffset(const void* p) {
  return (unsigned)(unsigned long long)p;
}

// One async 16B-per-lane copy (512B per wave-issue), tracked by ASYNCcnt
__device__ __forceinline__ void asyncCopy16(unsigned ldsOff, unsigned long long gaddr) {
  asm volatile("global_load_async_to_lds_b128 %0, %1, off"
               :: "v"(ldsOff), "v"(gaddr) : "memory");
}

#define WAIT_ASYNC_0  asm volatile("s_wait_asynccnt 0x0" ::: "memory")
#define WAIT_ASYNC_8  asm volatile("s_wait_asynccnt 0x8" ::: "memory")
#define WAIT_ASYNC_16 asm volatile("s_wait_asynccnt 0x10" ::: "memory")

// ---------------------------------------------------------------------------
// Prep: f32 -> bf16 conversion of activations; weights converted + transposed
// to [HD, C] so WMMA B-tiles are contiguous 32B per lane. Q scale folded in.
// ---------------------------------------------------------------------------
__global__ __launch_bounds__(256) void prep_x(const float* __restrict__ x,
                                              unsigned short* __restrict__ o, int n) {
  int i = blockIdx.x * 256 + threadIdx.x;
  if (i < n) o[i] = f2bf(x[i]);
}

__global__ __launch_bounds__(256) void prep_w(const float* __restrict__ w,
                                              unsigned short* __restrict__ wt,
                                              float scale) {
  int i = blockIdx.x * 256 + threadIdx.x;      // over C*HD (exact multiple)
  if (i < C * HD) {
    int n = i / C, c = i - n * C;
    wt[n * C + c] = f2bf(w[c * HD + n] * scale);
  }
}

// ---------------------------------------------------------------------------
// QKV GEMM: [8192 x 512] x [512 x 512] in bf16 WMMA, f32 accum.
// 8 waves/WG, each wave owns 16 rows x 64 cols (4 accumulators).
// mode 0: store [B,H,S,D] (Q,K). mode 1: store [B,H,D,S] (V transposed).
// ---------------------------------------------------------------------------
__global__ __launch_bounds__(256) void qkv_gemm(const unsigned short* __restrict__ X,
                                                const unsigned short* __restrict__ WT,
                                                unsigned short* __restrict__ OUT,
                                                int mode) {
  const int w  = threadIdx.x >> 5;
  const int l  = threadIdx.x & 31;
  const int lh = l & 15, lg = l >> 4;
  const int r0 = blockIdx.x * 128 + w * 16;
  const int n0 = blockIdx.y * 64;

  const unsigned short* Arow = X + (size_t)(r0 + lh) * C;
  v8f acc[4];
#pragma unroll
  for (int j = 0; j < 4; ++j) acc[j] = v8f{};

  for (int kc = 0; kc < C; kc += 32) {
    v16bf a = *(const v16bf*)(Arow + kc + lg * 16);
    const unsigned short* wb = WT + (size_t)(n0 + lh) * C + kc + lg * 16;
    if (kc + 32 < C) __builtin_prefetch((const void*)(Arow + kc + 32), 0, 1);
#pragma unroll
    for (int j = 0; j < 4; ++j) {
      v16bf b = *(const v16bf*)(wb + (size_t)j * 16 * C);
      acc[j] = wmma_bf16(a, b, acc[j]);
    }
  }

#pragma unroll
  for (int j = 0; j < 4; ++j) {
#pragma unroll
    for (int v = 0; v < 8; ++v) {
      int r = r0 + 8 * lg + v;                 // global row (b*S + s)
      int n = n0 + j * 16 + lh;                // output column (h*64 + d)
      int b = r >> 11, s = r & (S - 1);
      int h = n >> 6,  d = n & 63;
      size_t o = (mode == 0)
                   ? ((size_t)((b * H + h) * S + s)) * D + d
                   : ((size_t)((b * H + h) * D + d)) * S + s;
      OUT[o] = f2bf(acc[j][v]);
    }
  }
}

// ---------------------------------------------------------------------------
// Fused talking-heads attention. One WG per (batch, 16-query tile).
// 8 waves = 8 heads. Two-pass streaming softmax. K/V tiles are staged into
// LDS with double-buffered GLOBAL_LOAD_ASYNC_TO_LDS_B128 (ASYNCcnt-tracked),
// overlapping DMA of tile t+1 with WMMA + softmax of tile t.
// ---------------------------------------------------------------------------
__global__ __launch_bounds__(256) void attn_fused(const unsigned short* __restrict__ Q,
                                                  const unsigned short* __restrict__ K,
                                                  const unsigned short* __restrict__ Vt,
                                                  const float* __restrict__ pre,
                                                  const float* __restrict__ post,
                                                  float* __restrict__ out) {
  __shared__ __align__(64) unsigned short ldsK[H][2][32 * 64];  // 64 KB: K tiles [32k][64d]
  __shared__ __align__(64) unsigned short ldsV[H][2][64 * 32];  // 64 KB: V tiles [64d][32k]
  __shared__ __align__(32) float          ldsL[H][16][32];      // 16 KB: logit / P exchange
  __shared__ __align__(32) unsigned short ldsP[H][16][32];      //  8 KB: bf16 A-layout staging

  const int w  = threadIdx.x >> 5;          // wave id == head id
  const int l  = threadIdx.x & 31;
  const int lh = l & 15, lg = l >> 4;
  const int q0 = blockIdx.x * 16;
  const int b  = blockIdx.y;

  const unsigned short* Qh = Q  + (size_t)(b * H + w) * S * D;
  const unsigned short* Kh = K  + (size_t)(b * H + w) * S * D;
  const unsigned short* Vh = Vt + (size_t)(b * H + w) * D * S;

  // Async stage: K tile (rows kt..kt+31 x D) is contiguous 4KB
  auto issueK = [&](int kt, int buf) {
    unsigned lbase = ldsOffset(&ldsK[w][buf][0]) + l * 16;
    unsigned long long g = (unsigned long long)(Kh + (size_t)kt * D) + l * 16;
#pragma unroll
    for (int i = 0; i < 8; ++i)
      asyncCopy16(lbase + i * 512, g + i * 512);
  };
  // Async stage: V tile (64 d-rows x 32 k) -> LDS [64][32], 64B per row
  auto issueV = [&](int kt, int buf) {
    int row0 = l >> 2;
    int bo   = (l & 3) * 16;
    unsigned lbase = ldsOffset(&ldsV[w][buf][0]);
#pragma unroll
    for (int i = 0; i < 8; ++i) {
      int r = i * 8 + row0;
      asyncCopy16(lbase + r * 64 + bo,
                  (unsigned long long)(Vh + (size_t)r * S + kt) + bo);
    }
  };

  // Q tile (16 q-rows x 64 d) resident as two WMMA A operands
  v16bf qa0 = *(const v16bf*)(Qh + (size_t)(q0 + lh) * D + lg * 16);
  v16bf qa1 = *(const v16bf*)(Qh + (size_t)(q0 + lh) * D + 32 + lg * 16);

  float mix_pre[H], mix_post[H];
#pragma unroll
  for (int h = 0; h < H; ++h) {
    mix_pre[h]  = pre[h * H + w];            // pre[h, i=w]
    mix_post[h] = post[h * H + w];           // post[i=h, h_out=w]
  }

  float mrun[8], lrun[8];
#pragma unroll
  for (int v = 0; v < 8; ++v) { mrun[v] = -1e30f; lrun[v] = 0.f; }

  // -------- pass 1: online max & sum of pre-mixed logits --------
  issueK(0, 0);
  for (int kt = 0; kt < S; kt += 32) {
    const int buf = (kt >> 5) & 1;
    if (kt + 32 < S) { issueK(kt + 32, buf ^ 1); WAIT_ASYNC_8; }
    else             { WAIT_ASYNC_0; }

    const unsigned short* kb = &ldsK[w][buf][0];
    v16bf bk00 = *(const v16bf*)(kb + lh * 64 + lg * 16);
    v16bf bk01 = *(const v16bf*)(kb + lh * 64 + 32 + lg * 16);
    v16bf bk10 = *(const v16bf*)(kb + (16 + lh) * 64 + lg * 16);
    v16bf bk11 = *(const v16bf*)(kb + (16 + lh) * 64 + 32 + lg * 16);
    v8f z = v8f{};
    v8f L0 = wmma_bf16(qa1, bk01, wmma_bf16(qa0, bk00, z));
    v8f L1 = wmma_bf16(qa1, bk11, wmma_bf16(qa0, bk10, z));

    __syncthreads();
#pragma unroll
    for (int v = 0; v < 8; ++v) {
      ldsL[w][v + 8 * lg][lh]      = L0[v];
      ldsL[w][v + 8 * lg][16 + lh] = L1[v];
    }
    __syncthreads();

#pragma unroll
    for (int v = 0; v < 8; ++v) {
      float s0 = 0.f, s1 = 0.f;
#pragma unroll
      for (int h = 0; h < H; ++h) {
        s0 += mix_pre[h] * ldsL[h][v + 8 * lg][lh];
        s1 += mix_pre[h] * ldsL[h][v + 8 * lg][16 + lh];
      }
      float mx = fmaxf(s0, s1);
#pragma unroll
      for (int off = 8; off >= 1; off >>= 1)
        mx = fmaxf(mx, __shfl_xor(mx, off, 32));
      float mnew = fmaxf(mrun[v], mx);
      float ssum = __expf(s0 - mnew) + __expf(s1 - mnew);
#pragma unroll
      for (int off = 8; off >= 1; off >>= 1)
        ssum += __shfl_xor(ssum, off, 32);
      lrun[v] = lrun[v] * __expf(mrun[v] - mnew) + ssum;
      mrun[v] = mnew;
    }
  }

  float rl[8];
#pragma unroll
  for (int v = 0; v < 8; ++v) rl[v] = 1.f / lrun[v];

  v8f oacc[4];
#pragma unroll
  for (int j = 0; j < 4; ++j) oacc[j] = v8f{};

  // -------- pass 2: recompute, normalize, post-mix, P@V --------
  issueK(0, 0);
  issueV(0, 0);
  for (int kt = 0; kt < S; kt += 32) {
    const int buf = (kt >> 5) & 1;
    if (kt + 32 < S) { issueK(kt + 32, buf ^ 1); issueV(kt + 32, buf ^ 1); WAIT_ASYNC_16; }
    else             { WAIT_ASYNC_0; }

    const unsigned short* kb = &ldsK[w][buf][0];
    v16bf bk00 = *(const v16bf*)(kb + lh * 64 + lg * 16);
    v16bf bk01 = *(const v16bf*)(kb + lh * 64 + 32 + lg * 16);
    v16bf bk10 = *(const v16bf*)(kb + (16 + lh) * 64 + lg * 16);
    v16bf bk11 = *(const v16bf*)(kb + (16 + lh) * 64 + 32 + lg * 16);
    v8f z = v8f{};
    v8f L0 = wmma_bf16(qa1, bk01, wmma_bf16(qa0, bk00, z));
    v8f L1 = wmma_bf16(qa1, bk11, wmma_bf16(qa0, bk10, z));

    __syncthreads();
#pragma unroll
    for (int v = 0; v < 8; ++v) {
      ldsL[w][v + 8 * lg][lh]      = L0[v];
      ldsL[w][v + 8 * lg][16 + lh] = L1[v];
    }
    __syncthreads();

    float P0[8], P1[8];
#pragma unroll
    for (int v = 0; v < 8; ++v) {
      float s0 = 0.f, s1 = 0.f;
#pragma unroll
      for (int h = 0; h < H; ++h) {
        s0 += mix_pre[h] * ldsL[h][v + 8 * lg][lh];
        s1 += mix_pre[h] * ldsL[h][v + 8 * lg][16 + lh];
      }
      P0[v] = __expf(s0 - mrun[v]) * rl[v];
      P1[v] = __expf(s1 - mrun[v]) * rl[v];
    }

    __syncthreads();
#pragma unroll
    for (int v = 0; v < 8; ++v) {
      ldsL[w][v + 8 * lg][lh]      = P0[v];
      ldsL[w][v + 8 * lg][16 + lh] = P1[v];
    }
    __syncthreads();

    // post-softmax head mix -> A (this wave's output head), stage as bf16 A-layout
#pragma unroll
    for (int v = 0; v < 8; ++v) {
      float a0 = 0.f, a1 = 0.f;
#pragma unroll
      for (int h = 0; h < H; ++h) {
        a0 += mix_post[h] * ldsL[h][v + 8 * lg][lh];
        a1 += mix_post[h] * ldsL[h][v + 8 * lg][16 + lh];
      }
      ldsP[w][v + 8 * lg][lh]      = f2bf(a0);
      ldsP[w][v + 8 * lg][16 + lh] = f2bf(a1);
    }
    __syncthreads();

    v16bf pa = *(const v16bf*)(&ldsP[w][lh][lg * 16]);   // A-layout: row lh, K 0..31
    const unsigned short* vb = &ldsV[w][buf][0];
#pragma unroll
    for (int j = 0; j < 4; ++j) {
      v16bf bv = *(const v16bf*)(vb + (j * 16 + lh) * 32 + lg * 16);
      oacc[j] = wmma_bf16(pa, bv, oacc[j]);
    }
  }

  // store output [B, S, H*D] f32
#pragma unroll
  for (int j = 0; j < 4; ++j) {
#pragma unroll
    for (int v = 0; v < 8; ++v) {
      int q = q0 + v + 8 * lg;
      out[((size_t)(b * S + q)) * HD + w * 64 + j * 16 + lh] = oacc[j][v];
    }
  }
}

// ---------------------------------------------------------------------------
extern "C" void kernel_launch(void* const* d_in, const int* in_sizes, int n_in,
                              void* d_out, int out_size, void* d_ws, size_t ws_size,
                              hipStream_t stream) {
  const float* x    = (const float*)d_in[0];
  const float* wq   = (const float*)d_in[1];
  const float* wk   = (const float*)d_in[2];
  const float* wv   = (const float*)d_in[3];
  const float* pre  = (const float*)d_in[4];
  const float* post = (const float*)d_in[5];

  char* ws = (char*)d_ws;
  const size_t xBytes = (size_t)ROWS * C * 2;           // 8 MB bf16 activations
  const size_t wBytes = (size_t)C * HD * 2;             // 512 KB per transposed weight
  const size_t qBytes = (size_t)BATCH * H * S * D * 2;  // 8 MB per Q/K/V

  unsigned short* X16 = (unsigned short*)(ws);
  unsigned short* WqT = (unsigned short*)(ws + xBytes);
  unsigned short* WkT = (unsigned short*)(ws + xBytes + wBytes);
  unsigned short* WvT = (unsigned short*)(ws + xBytes + 2 * wBytes);
  unsigned short* Qb  = (unsigned short*)(ws + xBytes + 3 * wBytes);
  unsigned short* Kb  = (unsigned short*)(ws + xBytes + 3 * wBytes + qBytes);
  unsigned short* Vb  = (unsigned short*)(ws + xBytes + 3 * wBytes + 2 * qBytes);

  prep_x<<<(ROWS * C + 255) / 256, 256, 0, stream>>>(x, X16, ROWS * C);
  prep_w<<<(C * HD + 255) / 256, 256, 0, stream>>>(wq, WqT, 0.125f);  // 1/sqrt(64)
  prep_w<<<(C * HD + 255) / 256, 256, 0, stream>>>(wk, WkT, 1.0f);
  prep_w<<<(C * HD + 255) / 256, 256, 0, stream>>>(wv, WvT, 1.0f);

  dim3 ggrid(ROWS / 128, HD / 64);
  qkv_gemm<<<ggrid, 256, 0, stream>>>(X16, WqT, Qb, 0);
  qkv_gemm<<<ggrid, 256, 0, stream>>>(X16, WkT, Kb, 0);
  qkv_gemm<<<ggrid, 256, 0, stream>>>(X16, WvT, Vb, 1);

  attn_fused<<<dim3(S / 16, BATCH), 256, 0, stream>>>(Qb, Kb, Vb, pre, post,
                                                      (float*)d_out);
}
